// FramePrimerDecoder_79783312490669
// MI455X (gfx1250) — matrix operand order repaired and checked by reading, not compiled, not run on hardware
//
#include <hip/hip_runtime.h>

typedef __bf16 bf16;
typedef __attribute__((ext_vector_type(16))) __bf16 bf16x16;
typedef __attribute__((ext_vector_type(8))) float floatx8;

#define BINS_ 1024
#define W_    1024
#define B_    4
#define H_    4
#define HD_   256
#define FF_   2048
#define MTOK_ 4096   // B_*W_

// ---------------- WMMA GEMM: C[M,N] = A[M,K] * op(B) (+bias, epilogue) ----------------
// NT: B stored row-major [N][K] (weights / K-rows / Q-rows). NN: B row-major [K][N].
#define BM 128
#define BN 64
#define BK 32
#define APAD 40   // LDS row stride (elements): 80B -> 16B-aligned b128, conflict-free pair reads

union FragBF { bf16x16 v; unsigned int u[8]; };
union Seg128 { uint4 q; unsigned short s[8]; };

// CDNA5 async global->LDS copy (ASYNCcnt-tracked). LDS address = low 32 bits of the
// generic pointer (flat LDS aperture, wave-relative).
__device__ __forceinline__ void async_copy_b128(const bf16* g, unsigned short* l) {
  const unsigned loff = (unsigned)(unsigned long long)(void*)l;
  asm volatile("global_load_async_to_lds_b128 %0, %1, off"
               :: "v"(loff), "v"((unsigned long long)g) : "memory");
}
__device__ __forceinline__ void wait_async() {
  asm volatile("s_wait_asynccnt 0x0" ::: "memory");
  asm volatile("s_wait_xcnt 0x0" ::: "memory");
}

template<bool NT, bool OUT_BF16, bool RESID, bool RELUSQ, bool COPYB>
__global__ __launch_bounds__(256) void gemm_wmma(
    const bf16* __restrict__ A, int lda, long sAb, long sAh,
    const bf16* __restrict__ Bm, int ldb, long sBb, long sBh,
    float* __restrict__ Df, bf16* __restrict__ Dh, int ldd, long sDb, long sDh,
    const float* __restrict__ bias, bf16* __restrict__ copyOut,
    int M, int N, int K, int Hdiv)
{
  __shared__ unsigned short As[2][BM * APAD];
  __shared__ unsigned short Bs[2][BN * APAD];

  const int z  = blockIdx.z;
  const int zb = z / Hdiv, zh = z % Hdiv;
  A  += (size_t)zb * sAb + (size_t)zh * sAh;
  Bm += (size_t)zb * sBb + (size_t)zh * sBh;
  const size_t dOff = (size_t)zb * sDb + (size_t)zh * sDh;

  const int tid  = threadIdx.x;
  const int lane = tid & 31;
  const int wv   = tid >> 5;
  const int lm   = lane & 15;
  const int lhi  = lane >> 4;
  const int mW   = (wv >> 1) * 32;  // wave m-base in block: 0/32/64/96
  const int nW   = (wv & 1) * 32;   // wave n-base in block: 0/32

  const int mBlock = blockIdx.y * BM;
  const int nBlock = blockIdx.x * BN;

  floatx8 acc[4];
  #pragma unroll
  for (int t = 0; t < 4; ++t)
    #pragma unroll
    for (int j = 0; j < 8; ++j) acc[t][j] = 0.0f;

  const int rowA = tid >> 2, segA = tid & 3;   // 64 rows x 4 b128 segments

  auto stage = [&](int kt, int b) {
    // A tile: 128 rows x 32 cols bf16 (two async b128 per thread)
    async_copy_b128(A + (size_t)(mBlock + rowA) * lda + kt + segA * 8,
                    &As[b][rowA * APAD + segA * 8]);
    async_copy_b128(A + (size_t)(mBlock + rowA + 64) * lda + kt + segA * 8,
                    &As[b][(rowA + 64) * APAD + segA * 8]);
    if (NT) {       // B tile: 64 n-rows x 32 k (async)
      async_copy_b128(Bm + (size_t)(nBlock + rowA) * ldb + kt + segA * 8,
                      &Bs[b][rowA * APAD + segA * 8]);
    } else {        // B tile [K][N]: sync load rows, transpose-scatter into Bs[n][k]
      const int krow = tid >> 3, seg = tid & 7;
      Seg128 t4;
      t4.q = *reinterpret_cast<const uint4*>(Bm + (size_t)(kt + krow) * ldb + nBlock + seg * 8);
      #pragma unroll
      for (int j = 0; j < 8; ++j) Bs[b][(seg * 8 + j) * APAD + krow] = t4.s[j];
    }
  };

  // B fragment (32x16): lanes 0-15 K={0..7,16..23}, lanes 16-31 K={8..15,24..31}, 2 K per VGPR
  auto loadB = [&](int b, int n0, FragBF& f) {
    #pragma unroll
    for (int i = 0; i < 8; ++i)
      f.u[i] = *reinterpret_cast<const unsigned int*>(
          &Bs[b][(n0 + lm) * APAD + (i >> 2) * 16 + lhi * 8 + (i & 3) * 2]);
  };
  // A fragment (16x32): lanes 0-15 K={0..7,16..23}, lanes 16-31 K={8..15,24..31}
  auto loadA = [&](int b, int m0, FragBF& f) {
    #pragma unroll
    for (int i = 0; i < 8; ++i)
      f.u[i] = *reinterpret_cast<const unsigned int*>(
          &As[b][(m0 + lm) * APAD + (i >> 2) * 16 + lhi * 8 + (i & 3) * 2]);
  };

  stage(0, 0);
  wait_async();
  __syncthreads();

  int cur = 0;
  for (int kt = 0; kt < K; kt += BK) {
    if (kt + BK < K) stage(kt + BK, cur ^ 1);   // prefetch next tile (overlaps WMMA)

    FragBF a0, a1, b0, b1;
    loadA(cur, mW, a0);      loadA(cur, mW + 16, a1);
    loadB(cur, nW, b0);      loadB(cur, nW + 16, b1);
    acc[0] = __builtin_amdgcn_wmma_f32_16x16x32_bf16(false, a0.v, false, b0.v, (short)0, acc[0], false, false);
    acc[1] = __builtin_amdgcn_wmma_f32_16x16x32_bf16(false, a0.v, false, b1.v, (short)0, acc[1], false, false);
    acc[2] = __builtin_amdgcn_wmma_f32_16x16x32_bf16(false, a1.v, false, b0.v, (short)0, acc[2], false, false);
    acc[3] = __builtin_amdgcn_wmma_f32_16x16x32_bf16(false, a1.v, false, b1.v, (short)0, acc[3], false, false);

    wait_async();
    __syncthreads();
    cur ^= 1;
  }

  // epilogue: C layout per doc (VGPR r: lanes 0-15 -> M=r, lanes 16-31 -> M=r+8)
  #pragma unroll
  for (int t = 0; t < 4; ++t) {
    const int n = nBlock + nW + (t & 1) * 16 + lm;
    const float bv = bias ? bias[n] : 0.0f;
    #pragma unroll
    for (int r = 0; r < 8; ++r) {
      const int m = mBlock + mW + (t >> 1) * 16 + lhi * 8 + r;
      const size_t idx = dOff + (size_t)m * ldd + n;
      float v = acc[t][r] + bv;
      if (RELUSQ) { v = v > 0.0f ? v : 0.0f; v = v * v; }
      if (RESID)  v += Df[idx];
      if (OUT_BF16) Dh[idx] = (bf16)v; else Df[idx] = v;
      if (COPYB)  copyOut[idx] = (bf16)v;
    }
  }
}

// ---------------- elementwise / reduction kernels ----------------
__global__ __launch_bounds__(256) void cvt_bf16_kernel(const float* __restrict__ in,
                                                       bf16* __restrict__ out, int n) {
  int t = blockIdx.x * 256 + threadIdx.x;
  if (t < n) out[t] = (bf16)in[t];
}

__global__ __launch_bounds__(256) void er_transpose_kernel(const float* __restrict__ er,
                                                           bf16* __restrict__ out) {
  int t = blockIdx.x * 256 + threadIdx.x;   // out[w][d] = er[d][w]
  int d = t % HD_, w = t / HD_;
  out[t] = (bf16)er[(size_t)d * W_ + w];
}

__global__ __launch_bounds__(256) void in_project_kernel(
    const float* __restrict__ x, const float* __restrict__ skip,
    const float* __restrict__ inW, const float* __restrict__ skW,
    float* __restrict__ xw, bf16* __restrict__ swb) {
  size_t t = (size_t)blockIdx.x * 256 + threadIdx.x;  // over B*BINS*W, w fastest
  size_t w = t % W_;
  size_t d = (t / W_) % BINS_;
  size_t b = t / ((size_t)W_ * BINS_);
  const size_t cs = (size_t)BINS_ * W_;
  const float* xp = x + b * 8 * cs + d * W_ + w;
  const float* sp = skip + b * 8 * cs + d * W_ + w;
  float ax = 0.0f, as = 0.0f;
  #pragma unroll
  for (int c = 0; c < 8; ++c) { ax += xp[c * cs] * inW[c]; as += sp[c * cs] * skW[c]; }
  size_t o = b * (size_t)W_ * BINS_ + w * BINS_ + d;
  xw[o] = ax; swb[o] = (bf16)as;
}

template<bool OUT_BF16>
__global__ __launch_bounds__(256) void layernorm_kernel(
    const float* __restrict__ X, const float* __restrict__ g, const float* __restrict__ b,
    bf16* __restrict__ outH, float* __restrict__ outF) {
  __shared__ float red[256];
  const int row = blockIdx.x;
  const float* xr = X + (size_t)row * BINS_;
  float v[4]; float s = 0.0f;
  #pragma unroll
  for (int i = 0; i < 4; ++i) { v[i] = xr[threadIdx.x + i * 256]; s += v[i]; }
  red[threadIdx.x] = s; __syncthreads();
  for (int st = 128; st > 0; st >>= 1) {
    if (threadIdx.x < st) red[threadIdx.x] += red[threadIdx.x + st];
    __syncthreads();
  }
  const float mean = red[0] * (1.0f / BINS_);
  __syncthreads();
  float s2 = 0.0f;
  #pragma unroll
  for (int i = 0; i < 4; ++i) { float d = v[i] - mean; s2 += d * d; }
  red[threadIdx.x] = s2; __syncthreads();
  for (int st = 128; st > 0; st >>= 1) {
    if (threadIdx.x < st) red[threadIdx.x] += red[threadIdx.x + st];
    __syncthreads();
  }
  const float rstd = rsqrtf(red[0] * (1.0f / BINS_) + 1e-5f);
  #pragma unroll
  for (int i = 0; i < 4; ++i) {
    const int c = threadIdx.x + i * 256;
    const float y = (v[i] - mean) * rstd * g[c] + b[c];
    if (OUT_BF16) outH[(size_t)row * BINS_ + c] = (bf16)y;
    else          outF[(size_t)row * BINS_ + c] = y;
  }
}

__global__ __launch_bounds__(256) void dwconv_kernel(
    const bf16* __restrict__ in, const float* __restrict__ Kc,
    const float* __restrict__ bK, bf16* __restrict__ out) {
  size_t t = (size_t)blockIdx.x * 256 + threadIdx.x;  // [b][w][c], c fastest
  const int c = t % BINS_;
  const int w = (t / BINS_) % W_;
  float acc = bK[c] + Kc[c * 3 + 1] * (float)in[t];
  if (w > 0)      acc += Kc[c * 3 + 0] * (float)in[t - BINS_];
  if (w < W_ - 1) acc += Kc[c * 3 + 2] * (float)in[t + BINS_];
  out[t] = (bf16)acc;
}

// softmax over rows of S (f32, width W_), scale 1/32, bf16 written in-place over the row
__global__ __launch_bounds__(256) void softmax_kernel(float* __restrict__ S) {
  __shared__ float red[256];
  float* sr = S + (size_t)blockIdx.x * W_;
  float v[4]; float mx = -3.4e38f;
  #pragma unroll
  for (int i = 0; i < 4; ++i) { v[i] = sr[threadIdx.x + i * 256] * 0.03125f; mx = fmaxf(mx, v[i]); }
  red[threadIdx.x] = mx; __syncthreads();
  for (int st = 128; st > 0; st >>= 1) {
    if (threadIdx.x < st) red[threadIdx.x] = fmaxf(red[threadIdx.x], red[threadIdx.x + st]);
    __syncthreads();
  }
  mx = red[0]; __syncthreads();
  float s = 0.0f;
  #pragma unroll
  for (int i = 0; i < 4; ++i) { v[i] = __expf(v[i] - mx); s += v[i]; }
  red[threadIdx.x] = s; __syncthreads();
  for (int st = 128; st > 0; st >>= 1) {
    if (threadIdx.x < st) red[threadIdx.x] += red[threadIdx.x + st];
    __syncthreads();
  }
  const float inv = 1.0f / red[0];
  __syncthreads();  // all f32 reads done before bf16 overwrite
  bf16* pr = reinterpret_cast<bf16*>(sr);
  #pragma unroll
  for (int i = 0; i < 4; ++i) pr[threadIdx.x + i * 256] = (bf16)(v[i] * inv);
}

__global__ __launch_bounds__(256) void transpose_kernel(const float* __restrict__ in,
                                                        float* __restrict__ out) {
  __shared__ float tile[32][33];
  const int b = blockIdx.z, w0 = blockIdx.x * 32, d0 = blockIdx.y * 32;
  const int tx = threadIdx.x & 31, ty = threadIdx.x >> 5;
  const float* ip = in + (size_t)b * W_ * BINS_;
  #pragma unroll
  for (int j = 0; j < 4; ++j)
    tile[ty + j * 8][tx] = ip[(size_t)(w0 + ty + j * 8) * BINS_ + d0 + tx];
  __syncthreads();
  float* op = out + (size_t)b * BINS_ * W_;
  #pragma unroll
  for (int j = 0; j < 4; ++j)
    op[(size_t)(d0 + ty + j * 8) * W_ + w0 + tx] = tile[tx][ty + j * 8];
}

// ---------------- host-side attention pass ----------------
static void attention_pass(
    const bf16* qin, const bf16* kvin,
    const bf16* wq, const float* qb, const float* qK, const float* qKb,
    const bf16* wk, const float* kb, const float* kK, const float* kKb,
    const bf16* wv, const float* vb, const float* vK, const float* vKb,
    const bf16* wo, const float* ob, const bf16* ec,
    bf16* raw, bf16* qbuf, bf16* kbuf, bf16* vbuf,
    float* S, bf16* obuf, float* xw, bf16* xwbCopy, hipStream_t stream)
{
  const dim3 gProj(BINS_ / BN, MTOK_ / BM, 1);
  const dim3 gConv(MTOK_ * BINS_ / 256);
  // projections + depthwise conv
  gemm_wmma<true,true,false,false,false><<<gProj,256,0,stream>>>(
      qin,BINS_,0,0, wq,BINS_,0,0, nullptr, raw, BINS_,0,0, qb, nullptr, MTOK_,BINS_,BINS_,1);
  dwconv_kernel<<<gConv,256,0,stream>>>(raw, qK, qKb, qbuf);
  gemm_wmma<true,true,false,false,false><<<gProj,256,0,stream>>>(
      kvin,BINS_,0,0, wk,BINS_,0,0, nullptr, raw, BINS_,0,0, kb, nullptr, MTOK_,BINS_,BINS_,1);
  dwconv_kernel<<<gConv,256,0,stream>>>(raw, kK, kKb, kbuf);
  gemm_wmma<true,true,false,false,false><<<gProj,256,0,stream>>>(
      kvin,BINS_,0,0, wv,BINS_,0,0, nullptr, raw, BINS_,0,0, vb, nullptr, MTOK_,BINS_,BINS_,1);
  dwconv_kernel<<<gConv,256,0,stream>>>(raw, vK, vKb, vbuf);
  // scores: S = Q K^T ; then S += Ecol Q^T   (softmax applies 1/32)
  const dim3 gS(W_ / BN, W_ / BM, B_ * H_);
  gemm_wmma<true,false,false,false,false><<<gS,256,0,stream>>>(
      qbuf,BINS_,(long)W_*BINS_,(long)HD_, kbuf,BINS_,(long)W_*BINS_,(long)HD_,
      S, nullptr, W_, (long)H_*W_*W_, (long)W_*W_, nullptr, nullptr, W_,W_,HD_, H_);
  gemm_wmma<true,false,true,false,false><<<gS,256,0,stream>>>(
      ec,HD_,0,0, qbuf,BINS_,(long)W_*BINS_,(long)HD_,
      S, nullptr, W_, (long)H_*W_*W_, (long)W_*W_, nullptr, nullptr, W_,W_,HD_, H_);
  softmax_kernel<<<dim3(B_ * H_ * W_),256,0,stream>>>(S);
  // O = P @ V  (P bf16 in-place in S, row stride 2*W_ elements)
  const dim3 gO(HD_ / BN, W_ / BM, B_ * H_);
  gemm_wmma<false,true,false,false,false><<<gO,256,0,stream>>>(
      (const bf16*)S, 2*W_, (long)H_*W_*2*W_, (long)W_*2*W_,
      vbuf, BINS_, (long)W_*BINS_, (long)HD_,
      nullptr, obuf, BINS_, (long)W_*BINS_, (long)HD_, nullptr, nullptr, W_, HD_, W_, H_);
  // out projection + residual into xw
  if (xwbCopy)
    gemm_wmma<true,false,true,false,true><<<gProj,256,0,stream>>>(
        obuf,BINS_,0,0, wo,BINS_,0,0, xw, nullptr, BINS_,0,0, ob, xwbCopy, MTOK_,BINS_,BINS_,1);
  else
    gemm_wmma<true,false,true,false,false><<<gProj,256,0,stream>>>(
        obuf,BINS_,0,0, wo,BINS_,0,0, xw, nullptr, BINS_,0,0, ob, nullptr, MTOK_,BINS_,BINS_,1);
}

extern "C" void kernel_launch(void* const* d_in, const int* in_sizes, int n_in,
                              void* d_out, int out_size, void* d_ws, size_t ws_size,
                              hipStream_t stream) {
  (void)in_sizes; (void)n_in; (void)out_size; (void)ws_size;
  auto P = [&](int i) -> const float* { return (const float*)d_in[i]; };
  const float* x    = P(0);
  const float* skip = P(1);
  const float* inW  = P(2);
  const float* skW  = P(3);
  const int A1 = 4, A2 = 19; // qW,qb,qK,qKb, kW,kb,kK,kKb, vW,vb,vK,vKb, oW,ob,er
  const float* n1g = P(34); const float* n1b = P(35);
  const float* n2g = P(36); const float* n2b = P(37);
  const float* n3g = P(38); const float* n3b = P(39);
  const float* l1W = P(40); const float* l1b = P(41);
  const float* l2W = P(42); const float* l2b = P(43);

  // ---- workspace layout ----
  char* base = (char*)d_ws; size_t off = 0;
  auto take = [&](size_t bytes) -> char* {
    char* p = base + off; off = (off + bytes + 255) & ~(size_t)255; return p;
  };
  float* xw   = (float*)take((size_t)MTOK_ * BINS_ * 4);
  bf16*  swb  = (bf16*) take((size_t)MTOK_ * BINS_ * 2);
  bf16*  hb   = (bf16*) take((size_t)MTOK_ * BINS_ * 2);
  bf16*  raw  = (bf16*) take((size_t)MTOK_ * BINS_ * 2);
  bf16*  qbuf = (bf16*) take((size_t)MTOK_ * BINS_ * 2);
  bf16*  kbuf = (bf16*) take((size_t)MTOK_ * BINS_ * 2);
  bf16*  vbuf = (bf16*) take((size_t)MTOK_ * BINS_ * 2);
  char*  Sreg = take((size_t)B_ * H_ * W_ * W_ * 4);   // 64 MB; later reused for ffh/lnf
  float* S    = (float*)Sreg;
  bf16*  ffh  = (bf16*)Sreg;                            // [MTOK_, FF_] bf16 (16 MB)
  float* lnf  = (float*)(Sreg + (size_t)MTOK_ * FF_ * 2);
  bf16*  obuf = (bf16*) take((size_t)MTOK_ * BINS_ * 2);
  bf16*  xwb  = (bf16*) take((size_t)MTOK_ * BINS_ * 2);
  bf16* wq1 = (bf16*)take((size_t)1024*1024*2); bf16* wk1 = (bf16*)take((size_t)1024*1024*2);
  bf16* wv1 = (bf16*)take((size_t)1024*1024*2); bf16* wo1 = (bf16*)take((size_t)1024*1024*2);
  bf16* wq2 = (bf16*)take((size_t)1024*1024*2); bf16* wk2 = (bf16*)take((size_t)1024*1024*2);
  bf16* wv2 = (bf16*)take((size_t)1024*1024*2); bf16* wo2 = (bf16*)take((size_t)1024*1024*2);
  bf16* wl1 = (bf16*)take((size_t)FF_*BINS_*2); bf16* wl2 = (bf16*)take((size_t)BINS_*FF_*2);
  bf16* ec1 = (bf16*)take((size_t)W_*HD_*2);    bf16* ec2 = (bf16*)take((size_t)W_*HD_*2);

  // ---- weight prep (f32 -> bf16, er -> Ecol = er^T) ----
  auto cvt = [&](const float* src, bf16* dst, int n) {
    cvt_bf16_kernel<<<dim3((n + 255) / 256), 256, 0, stream>>>(src, dst, n);
  };
  cvt(P(A1+0), wq1, 1024*1024); cvt(P(A1+4), wk1, 1024*1024);
  cvt(P(A1+8), wv1, 1024*1024); cvt(P(A1+12), wo1, 1024*1024);
  cvt(P(A2+0), wq2, 1024*1024); cvt(P(A2+4), wk2, 1024*1024);
  cvt(P(A2+8), wv2, 1024*1024); cvt(P(A2+12), wo2, 1024*1024);
  cvt(l1W, wl1, FF_*BINS_);     cvt(l2W, wl2, BINS_*FF_);
  er_transpose_kernel<<<dim3(W_*HD_/256),256,0,stream>>>(P(A1+14), ec1);
  er_transpose_kernel<<<dim3(W_*HD_/256),256,0,stream>>>(P(A2+14), ec2);

  // ---- input projection ----
  in_project_kernel<<<dim3(MTOK_*BINS_/256),256,0,stream>>>(x, skip, inW, skW, xw, swb);

  // ---- attention 1 (self, mem = h) ----
  layernorm_kernel<true><<<dim3(MTOK_),256,0,stream>>>(xw, n1g, n1b, hb, nullptr);
  attention_pass(hb, hb,
      wq1, P(A1+1), P(A1+2), P(A1+3), wk1, P(A1+5), P(A1+6), P(A1+7),
      wv1, P(A1+9), P(A1+10), P(A1+11), wo1, P(A1+13), ec1,
      raw, qbuf, kbuf, vbuf, S, obuf, xw, nullptr, stream);

  // ---- attention 2 (cross, mem = sw) ----
  layernorm_kernel<true><<<dim3(MTOK_),256,0,stream>>>(xw, n2g, n2b, hb, nullptr);
  attention_pass(hb, swb,
      wq2, P(A2+1), P(A2+2), P(A2+3), wk2, P(A2+5), P(A2+6), P(A2+7),
      wv2, P(A2+9), P(A2+10), P(A2+11), wo2, P(A2+13), ec2,
      raw, qbuf, kbuf, vbuf, S, obuf, xw, xwb, stream);

  // ---- feed-forward: relu(xw@l1W^T+l1b)^2 @ l2W^T + l2b, residual into xw ----
  gemm_wmma<true,true,false,true,false><<<dim3(FF_/BN, MTOK_/BM, 1),256,0,stream>>>(
      xwb,BINS_,0,0, wl1,BINS_,0,0, nullptr, ffh, FF_,0,0, l1b, nullptr, MTOK_,FF_,BINS_,1);
  gemm_wmma<true,false,true,false,false><<<dim3(BINS_/BN, MTOK_/BM, 1),256,0,stream>>>(
      ffh,FF_,0,0, wl2,FF_,0,0, xw, nullptr, BINS_,0,0, l2b, nullptr, MTOK_,BINS_,FF_,1);

  // ---- final layernorm + transpose to [B,1,BINS,W] ----
  layernorm_kernel<false><<<dim3(MTOK_),256,0,stream>>>(xw, n3g, n3b, nullptr, lnf);
  transpose_kernel<<<dim3(W_/32, BINS_/32, B_),256,0,stream>>>(lnf, (float*)d_out);
}
